// MaskRCNN_37795712204840
// MI455X (gfx1250) — compile-verified, matrix operand order checked by reading, not run on hardware
//
#include <hip/hip_runtime.h>

// ---------------------------------------------------------------------------
// Mask R-CNN head for MI455X (gfx1250): RoIAlign + WMMA MLPs + implicit-GEMM
// convs + fused deconv/predictor.  All heavy math via v_wmma_f32_16x16x32_f16.
// B matrices stored K-panel-packed so every fragment is two b128 loads.
// ---------------------------------------------------------------------------

typedef __attribute__((ext_vector_type(16))) _Float16 v16h;
typedef __attribute__((ext_vector_type(8)))  _Float16 v8h;
typedef __attribute__((ext_vector_type(8)))  float    v8f;

#define NROI   512
#define CCH    256
#define FH     200
#define FW     336
#define DIN    12544        // 256*7*7
#define NCLS   81

// ---------------- WMMA fragment loaders (ISA 7.12.2 layouts) ----------------
// Fragment element pattern per lane: e<8 -> K = k0 + 8*(lane>>4) + e
//                                    e>=8 -> K = k0 + 16 + 8*(lane>>4) + (e-8)
// With 32 contiguous K-halves at `base`, that is two aligned 16-byte chunks.
__device__ __forceinline__ v16h frag_ld32(const _Float16* base, int lane) {
  const v8h* q = (const v8h*)(base + ((lane >> 4) << 3));
  v8h lo = q[0];
  v8h hi = q[2];   // +16 halves
  return __builtin_shufflevector(lo, hi, 0, 1, 2, 3, 4, 5, 6, 7,
                                 8, 9, 10, 11, 12, 13, 14, 15);
}
// A row-major [M,K]: lane -> row m0 + (lane&15), 32 contiguous K
__device__ __forceinline__ v16h frag_a_g(const _Float16* __restrict__ A, int lda,
                                         int m0, int k0, int lane) {
  return frag_ld32(A + (size_t)(m0 + (lane & 15)) * lda + k0, lane);
}
__device__ __forceinline__ v16h frag_a_lds(const _Float16* L, int ld, int k0, int lane) {
  return frag_ld32(L + (lane & 15) * ld + k0, lane);
}
// B panel-packed [K/32][Npad][32]: lane -> column n0 + (lane&15), contiguous K panel
__device__ __forceinline__ v16h frag_b_pk(const _Float16* __restrict__ Bt, int Npad,
                                          int k0, int n0, int lane) {
  return frag_ld32(Bt + ((size_t)(k0 >> 5) * Npad + n0 + (lane & 15)) * 32, lane);
}

#define WMMA_F16(a, b, c) \
  __builtin_amdgcn_wmma_f32_16x16x32_f16(false, (a), false, (b), (short)0, (c), false, false)

// --------------------------------- RoIAlign ---------------------------------
// mode 0: out[n*12544 + c*os*os + pos]  (flattened, for MLP GEMM A-matrix)
// mode 1: out[(n*256+c)*196 + pos]      (NC S layout for conv branch)
__global__ void roi_align_f16_kernel(const float* __restrict__ feat,
                                     const float* __restrict__ boxes,
                                     const int* __restrict__ bidx,
                                     _Float16* __restrict__ out,
                                     int os, int mode) {
  int n = blockIdx.x;
  float bx1 = boxes[n * 4 + 0], by1 = boxes[n * 4 + 1];
  float bx2 = boxes[n * 4 + 2], by2 = boxes[n * 4 + 3];
  int b = bidx[n];
  int P = os * 2;
  float invP = 1.0f / (float)P;
  int oss = os * os;
  int total = CCH * oss;
  for (int e = threadIdx.x; e < total; e += blockDim.x) {
    int c = e / oss;
    int pos = e - c * oss;
    int oy = pos / os, ox = pos - oy * os;
    const float* fb = feat + ((size_t)(b * CCH + c)) * (FH * FW);
    float acc = 0.f;
#pragma unroll
    for (int sy = 0; sy < 2; ++sy) {
      int py = oy * 2 + sy;
      float ys = by1 + (py + 0.5f) * invP * (by2 - by1);
      ys = fminf(fmaxf(ys, 0.f), (float)(FH - 1));
      int y0 = (int)floorf(ys);
      int y1i = min(y0 + 1, FH - 1);
      float ly = ys - (float)y0;
#pragma unroll
      for (int sx = 0; sx < 2; ++sx) {
        int px = ox * 2 + sx;
        float xs = bx1 + (px + 0.5f) * invP * (bx2 - bx1);
        xs = fminf(fmaxf(xs, 0.f), (float)(FW - 1));
        int x0 = (int)floorf(xs);
        int x1i = min(x0 + 1, FW - 1);
        float lx = xs - (float)x0;
        float f00 = fb[y0 * FW + x0],  f01 = fb[y0 * FW + x1i];
        float f10 = fb[y1i * FW + x0], f11 = fb[y1i * FW + x1i];
        acc += f00 * (1.f - ly) * (1.f - lx) + f01 * (1.f - ly) * lx +
               f10 * ly * (1.f - lx) + f11 * ly * lx;
      }
    }
    acc *= 0.25f;
    if (mode == 0) out[(size_t)n * DIN + c * oss + pos] = (_Float16)acc;
    else           out[((size_t)(n * CCH + c)) * 196 + pos] = (_Float16)acc;
  }
}

// --------------------------- Generic WMMA GEMM ------------------------------
// D[M,Npad] = A[M,K] @ Bpk ; one wave computes a 32x16 block (2 M-tiles reusing
// each B fragment), 8 waves/block.  f32out: float D, ld=realN, guard n<realN.
__global__ void gemm_wmma_kernel(const _Float16* __restrict__ A,
                                 const _Float16* __restrict__ Bt,
                                 void* __restrict__ D,
                                 int M, int K, int Npad, int realN,
                                 int relu, int f32out) {
  int lane = threadIdx.x & 31;
  int wave = threadIdx.x >> 5;
  int ntiles = Npad >> 4;
  int T = (M >> 5) * ntiles;           // 32-row M blocks
  int t = blockIdx.x * 8 + wave;
  if (t >= T) return;
  int mp = t / ntiles, nt = t - mp * ntiles;
  int m0 = mp << 5, n0 = nt << 4;
  v8f acc0 = {}, acc1 = {};
  for (int k0 = 0; k0 < K; k0 += 32) {
    v16h b  = frag_b_pk(Bt, Npad, k0, n0, lane);
    v16h a0 = frag_a_g(A, K, m0, k0, lane);
    acc0 = WMMA_F16(a0, b, acc0);
    v16h a1 = frag_a_g(A, K, m0 + 16, k0, lane);
    acc1 = WMMA_F16(a1, b, acc1);
  }
  int n = n0 + (lane & 15);
  int mb0 = m0 + ((lane >> 4) << 3);
  if (f32out) {
    float* Df = (float*)D;
    if (n < realN) {
#pragma unroll
      for (int v = 0; v < 8; ++v) {
        float x0 = acc0[v], x1 = acc1[v];
        if (relu) { x0 = fmaxf(x0, 0.f); x1 = fmaxf(x1, 0.f); }
        Df[(size_t)(mb0 + v) * realN + n] = x0;
        Df[(size_t)(mb0 + 16 + v) * realN + n] = x1;
      }
    }
  } else {
    _Float16* Dh = (_Float16*)D;
#pragma unroll
    for (int v = 0; v < 8; ++v) {
      float x0 = acc0[v], x1 = acc1[v];
      if (relu) { x0 = fmaxf(x0, 0.f); x1 = fmaxf(x1, 0.f); }
      Dh[(size_t)(mb0 + v) * Npad + n] = (_Float16)x0;
      Dh[(size_t)(mb0 + 16 + v) * Npad + n] = (_Float16)x1;
    }
  }
}

// ------------------- Conv3x3 (SAME) as implicit GEMM ------------------------
// X,Y: [512, 256, 196] f16.  Wc panel-packed [2304/32][256][32], k=ci*9+tap.
// Block = one (roi, 16-row spatial tile); 8 waves cover all 256 out channels.
#define CLDA 296   // 288 + pad (x2B = 592B, 16B-aligned rows)
__global__ void conv3x3_wmma_kernel(const _Float16* __restrict__ X,
                                    const _Float16* __restrict__ Wc,
                                    _Float16* __restrict__ Y) {
  __shared__ _Float16 lA[16 * CLDA];
  int n = blockIdx.x / 13;
  int mt = blockIdx.x - n * 13;
  int s0 = mt << 4;
  int lane = threadIdx.x & 31, wave = threadIdx.x >> 5;
  int co0a = wave << 5, co0b = co0a + 16;
  const _Float16* Xn = X + (size_t)n * CCH * 196;
  v8f acc0 = {}, acc1 = {};
  for (int cib = 0; cib < 8; ++cib) {        // 32 input channels per chunk
    for (int idx = threadIdx.x; idx < 16 * 288; idx += 256) {
      int row = idx / 288;
      int kk = idx - row * 288;
      int cil = kk / 9;
      int tap = kk - cil * 9;
      int dy = tap / 3 - 1, dx = tap - (tap / 3) * 3 - 1;
      int s = s0 + row;
      _Float16 val = (_Float16)0.f;
      if (s < 196) {
        int i = s / 14, j = s - (s / 14) * 14;
        int ii = i + dy, jj = j + dx;
        if (ii >= 0 && ii < 14 && jj >= 0 && jj < 14)
          val = Xn[(size_t)(cib * 32 + cil) * 196 + ii * 14 + jj];
      }
      lA[row * CLDA + kk] = val;
    }
    __syncthreads();
#pragma unroll
    for (int kk0 = 0; kk0 < 288; kk0 += 32) {
      v16h a = frag_a_lds(lA, CLDA, kk0, lane);
      v16h b0 = frag_b_pk(Wc, 256, cib * 288 + kk0, co0a, lane);
      acc0 = WMMA_F16(a, b0, acc0);
      v16h b1 = frag_b_pk(Wc, 256, cib * 288 + kk0, co0b, lane);
      acc1 = WMMA_F16(a, b1, acc1);
    }
    __syncthreads();
  }
  int sbase = s0 + ((lane >> 4) << 3);
  int coA = co0a + (lane & 15), coB = co0b + (lane & 15);
#pragma unroll
  for (int v = 0; v < 8; ++v) {
    int s = sbase + v;
    if (s < 196) {
      Y[((size_t)(n * CCH) + coA) * 196 + s] = (_Float16)fmaxf(acc0[v], 0.f);
      Y[((size_t)(n * CCH) + coB) * 196 + s] = (_Float16)fmaxf(acc1[v], 0.f);
    }
  }
}

// ------------- Fused deconv2x2(stride2)+relu -> 1x1 predictor ---------------
// X: [512,256,196] f16.  Wd: 4 panel-packed [256/32][256][32] per (dh,dw).
// Wp panel-packed [256/32][96][32].  Out: mask f32 [512, 81, 28, 28].
#define DLDA 264   // x2B = 528B, 16B-aligned rows
__global__ void deconv_mask_wmma_kernel(const _Float16* __restrict__ X,
                                        const _Float16* __restrict__ Wd,
                                        const _Float16* __restrict__ Wp,
                                        float* __restrict__ Out) {
  __shared__ _Float16 lA[16 * DLDA];
  __shared__ _Float16 lY[16 * DLDA];
  int idx = blockIdx.x;
  int n = idx / 52;
  int rem = idx - n * 52;
  int d = rem / 13;
  int mt = rem - d * 13;
  int di = d >> 1, dj = d & 1;
  int s0 = mt << 4;
  int lane = threadIdx.x & 31, wave = threadIdx.x >> 5;
  const _Float16* Xn = X + (size_t)n * CCH * 196;

  // stage A tile [16 rows x 256 ci]: thread = channel, contiguous-in-s reads
  {
    int ci = threadIdx.x;
#pragma unroll
    for (int row = 0; row < 16; ++row) {
      int s = s0 + row;
      lA[row * DLDA + ci] = (s < 196) ? Xn[(size_t)ci * 196 + s] : (_Float16)0.f;
    }
  }
  __syncthreads();

  // GEMM1: Y[16x256] = A @ Wd[d], relu, kept in LDS as f16
  const _Float16* Wdd = Wd + (size_t)d * 256 * 256;
  int co0a = wave << 5, co0b = co0a + 16;
  v8f acc0 = {}, acc1 = {};
#pragma unroll
  for (int k0 = 0; k0 < 256; k0 += 32) {
    v16h a = frag_a_lds(lA, DLDA, k0, lane);
    v16h b0 = frag_b_pk(Wdd, 256, k0, co0a, lane);
    acc0 = WMMA_F16(a, b0, acc0);
    v16h b1 = frag_b_pk(Wdd, 256, k0, co0b, lane);
    acc1 = WMMA_F16(a, b1, acc1);
  }
  {
    int coA = co0a + (lane & 15), coB = co0b + (lane & 15);
    int rbase = (lane >> 4) << 3;
#pragma unroll
    for (int v = 0; v < 8; ++v) {
      lY[(rbase + v) * DLDA + coA] = (_Float16)fmaxf(acc0[v], 0.f);
      lY[(rbase + v) * DLDA + coB] = (_Float16)fmaxf(acc1[v], 0.f);
    }
  }
  __syncthreads();

  // GEMM2: mask[16x96pad] = Y @ Wp (no relu); waves 0..5 cover 96 cols
  if (wave < 6) {
    int n0 = wave << 4;
    v8f acc = {};
#pragma unroll
    for (int k0 = 0; k0 < 256; k0 += 32) {
      v16h a = frag_a_lds(lY, DLDA, k0, lane);
      v16h b = frag_b_pk(Wp, 96, k0, n0, lane);
      acc = WMMA_F16(a, b, acc);
    }
    int kcls = n0 + (lane & 15);
    int rbase = (lane >> 4) << 3;
    if (kcls < NCLS) {
#pragma unroll
      for (int v = 0; v < 8; ++v) {
        int s = s0 + rbase + v;
        if (s < 196) {
          int i = s / 14, j = s - (s / 14) * 14;
          int p = 2 * i + di, q = 2 * j + dj;
          Out[((size_t)(n * NCLS) + kcls) * 784 + p * 28 + q] = acc[v];
        }
      }
    }
  }
}

// ----------------- weight conversions (all panel-packed) ---------------------
// f32 [K, realN] row-major -> f16 packed [K/32][padN][32] (zero-pad columns)
__global__ void cvt_pack_kernel(const float* __restrict__ s, _Float16* __restrict__ d,
                                int K, int realN, int padN) {
  int i = blockIdx.x * 256 + threadIdx.x;
  if (i >= K * padN) return;
  int kt = i / (padN * 32);
  int rem = i - kt * (padN * 32);
  int n = rem >> 5;
  int kk = rem & 31;
  int k = kt * 32 + kk;
  d[i] = (n < realN) ? (_Float16)s[(size_t)k * realN + n] : (_Float16)0.f;
}
// conv OIHW [256,256,3,3] -> packed [2304/32][256][32], k = ci*9 + tap
__global__ void cvt_convw_kernel(const float* __restrict__ m, _Float16* __restrict__ Wc) {
  int i = blockIdx.x * 256 + threadIdx.x;
  if (i >= 2304 * 256) return;
  int kt = i / (256 * 32);
  int rem = i - kt * (256 * 32);
  int co = rem >> 5;
  int kk = rem & 31;
  int k = kt * 32 + kk;
  int ci = k / 9, tap = k - ci * 9;
  Wc[i] = (_Float16)m[((size_t)co * 256 + ci) * 9 + tap];
}
// deconv IOHW [256,256,2,2] -> 4 packed [256/32][256][32] (k = ci, n = o)
__global__ void cvt_deconvw_kernel(const float* __restrict__ mdc, _Float16* __restrict__ Wd) {
  int i = blockIdx.x * 256 + threadIdx.x;
  if (i >= 4 * 65536) return;
  int d = i >> 16;
  int rem = i & 65535;
  int kt = rem / (256 * 32);
  int r2 = rem - kt * (256 * 32);
  int o = r2 >> 5;
  int kk = r2 & 31;
  int ci = kt * 32 + kk;
  Wd[i] = (_Float16)mdc[((size_t)ci * 256 + o) * 4 + d];
}
// mpr [81,256,1,1] -> packed [256/32][96][32] (k = o, n = class, zero-padded)
__global__ void cvt_mpr_kernel(const float* __restrict__ mpr, _Float16* __restrict__ Wp) {
  int i = blockIdx.x * 256 + threadIdx.x;
  if (i >= 256 * 96) return;
  int kt = i / (96 * 32);
  int rem = i - kt * (96 * 32);
  int n = rem >> 5;
  int kk = rem & 31;
  int o = kt * 32 + kk;
  Wp[i] = (n < NCLS) ? (_Float16)mpr[(size_t)n * 256 + o] : (_Float16)0.f;
}

// --------------------------------- launch -----------------------------------
extern "C" void kernel_launch(void* const* d_in, const int* in_sizes, int n_in,
                              void* d_out, int out_size, void* d_ws, size_t ws_size,
                              hipStream_t stream) {
  const float* feat  = (const float*)d_in[0];
  const float* boxes = (const float*)d_in[1];
  const float* bw1 = (const float*)d_in[2];
  const float* bw2 = (const float*)d_in[3];
  const float* bwo = (const float*)d_in[4];
  const float* cw1 = (const float*)d_in[5];
  const float* cw2 = (const float*)d_in[6];
  const float* cwo = (const float*)d_in[7];
  const float* m1  = (const float*)d_in[8];
  const float* m2  = (const float*)d_in[9];
  const float* m3  = (const float*)d_in[10];
  const float* m4  = (const float*)d_in[11];
  const float* mdc = (const float*)d_in[12];
  const float* mpr = (const float*)d_in[13];
  const int*   bidx = (const int*)d_in[14];
  float* out = (float*)d_out;

  char* ws = (char*)d_ws;
  size_t cur = 0;
  auto allocH = [&](size_t halves) -> _Float16* {
    _Float16* p = (_Float16*)(ws + cur);
    cur += ((halves * sizeof(_Float16) + 255) & ~(size_t)255);
    return p;
  };
  _Float16* A7   = allocH((size_t)NROI * DIN);        // 12.85 MB
  _Float16* Hb1  = allocH((size_t)NROI * 1024);
  _Float16* Hb2  = allocH((size_t)NROI * 1024);
  _Float16* Hc1  = allocH((size_t)NROI * 1024);
  _Float16* Hc2  = allocH((size_t)NROI * 1024);
  _Float16* bw1f = allocH((size_t)DIN * 1024);        // 25.7 MB
  _Float16* bw2f = allocH((size_t)1024 * 1024);
  _Float16* bwof = allocH((size_t)1024 * 336);        // 324 -> 336 pad
  _Float16* cw1f = allocH((size_t)DIN * 1024);
  _Float16* cw2f = allocH((size_t)1024 * 1024);
  _Float16* cwof = allocH((size_t)1024 * 96);         // 81 -> 96 pad
  _Float16* Wc1  = allocH((size_t)2304 * 256);
  _Float16* Wc2  = allocH((size_t)2304 * 256);
  _Float16* Wc3  = allocH((size_t)2304 * 256);
  _Float16* Wc4  = allocH((size_t)2304 * 256);
  _Float16* Wd   = allocH((size_t)4 * 256 * 256);
  _Float16* Wp   = allocH((size_t)256 * 96);
  _Float16* Xa   = allocH((size_t)NROI * CCH * 196);  // 51.4 MB
  _Float16* Xb   = allocH((size_t)NROI * CCH * 196);
  (void)ws_size; (void)in_sizes; (void)n_in; (void)out_size;

  auto g1 = [](int n) { return dim3((unsigned)((n + 255) / 256)); };

  // ---- weight conversions (panel-packed) ----
  cvt_pack_kernel<<<g1(DIN * 1024), 256, 0, stream>>>(bw1, bw1f, DIN, 1024, 1024);
  cvt_pack_kernel<<<g1(1024 * 1024), 256, 0, stream>>>(bw2, bw2f, 1024, 1024, 1024);
  cvt_pack_kernel<<<g1(1024 * 336), 256, 0, stream>>>(bwo, bwof, 1024, 324, 336);
  cvt_pack_kernel<<<g1(DIN * 1024), 256, 0, stream>>>(cw1, cw1f, DIN, 1024, 1024);
  cvt_pack_kernel<<<g1(1024 * 1024), 256, 0, stream>>>(cw2, cw2f, 1024, 1024, 1024);
  cvt_pack_kernel<<<g1(1024 * 96), 256, 0, stream>>>(cwo, cwof, 1024, NCLS, 96);
  cvt_convw_kernel<<<g1(2304 * 256), 256, 0, stream>>>(m1, Wc1);
  cvt_convw_kernel<<<g1(2304 * 256), 256, 0, stream>>>(m2, Wc2);
  cvt_convw_kernel<<<g1(2304 * 256), 256, 0, stream>>>(m3, Wc3);
  cvt_convw_kernel<<<g1(2304 * 256), 256, 0, stream>>>(m4, Wc4);
  cvt_deconvw_kernel<<<g1(4 * 65536), 256, 0, stream>>>(mdc, Wd);
  cvt_mpr_kernel<<<g1(256 * 96), 256, 0, stream>>>(mpr, Wp);

  // ---- RoIAlign (7x7 flat for MLPs, 14x14 for mask branch) ----
  roi_align_f16_kernel<<<dim3(NROI), 256, 0, stream>>>(feat, boxes, bidx, A7, 7, 0);
  roi_align_f16_kernel<<<dim3(NROI), 256, 0, stream>>>(feat, boxes, bidx, Xa, 14, 1);

  // ---- MLPs (WMMA GEMMs); outputs f32 into d_out ----
  auto gemm = [&](const _Float16* A, const _Float16* B, void* D,
                  int M, int K, int Npad, int realN, int relu, int f32o) {
    int tiles = (M >> 5) * (Npad >> 4);   // one wave per 32x16 block
    gemm_wmma_kernel<<<g1(tiles * 32), 256, 0, stream>>>(A, B, D, M, K, Npad, realN, relu, f32o);
  };
  float* box_out = out;                           // [512, 324]
  float* cls_out = out + (size_t)NROI * 324;      // [512, 81]
  float* msk_out = cls_out + (size_t)NROI * NCLS; // [512, 81, 28, 28]

  gemm(A7,  bw1f, Hb1, NROI, DIN,  1024, 1024, 1, 0);
  gemm(Hb1, bw2f, Hb2, NROI, 1024, 1024, 1024, 1, 0);
  gemm(Hb2, bwof, box_out, NROI, 1024, 336, 324, 0, 1);
  gemm(A7,  cw1f, Hc1, NROI, DIN,  1024, 1024, 1, 0);
  gemm(Hc1, cw2f, Hc2, NROI, 1024, 1024, 1024, 1, 0);
  gemm(Hc2, cwof, cls_out, NROI, 1024, 96, NCLS, 0, 1);

  // ---- mask branch: 4x conv3x3+relu (ping-pong Xa/Xb) ----
  dim3 cgrid(NROI * 13);
  conv3x3_wmma_kernel<<<cgrid, 256, 0, stream>>>(Xa, Wc1, Xb);
  conv3x3_wmma_kernel<<<cgrid, 256, 0, stream>>>(Xb, Wc2, Xa);
  conv3x3_wmma_kernel<<<cgrid, 256, 0, stream>>>(Xa, Wc3, Xb);
  conv3x3_wmma_kernel<<<cgrid, 256, 0, stream>>>(Xb, Wc4, Xa);

  // ---- fused deconv(2x2,s2)+relu -> 1x1 predictor ----
  deconv_mask_wmma_kernel<<<dim3(NROI * 4 * 13), 256, 0, stream>>>(Xa, Wd, Wp, msk_out);
}
